// EnhancedGenerator_24567212933410
// MI455X (gfx1250) — matrix operand order repaired and checked by reading, not compiled, not run on hardware
//
#include <hip/hip_runtime.h>
#include <hip/hip_bf16.h>
#include <math.h>

// ---------------------------------------------------------------------------
// MI455X / gfx1250 implementation. wave32, 256-thread blocks (8 waves).
// Trunk matmuls (conv3x3, 1x1 convs, window attention) use
// v_wmma_f32_16x16x32_bf16 (bf16 inputs, f32 accumulate).
// ---------------------------------------------------------------------------

typedef __attribute__((ext_vector_type(16))) __bf16 v16bf;
typedef __attribute__((ext_vector_type(8)))  float  v8f;

union FragAB {
  v16bf v;
  uint4 u[2];
  unsigned short s[16];
};

__device__ __forceinline__ unsigned short f2bf(float f) {
  unsigned int u = __builtin_bit_cast(unsigned int, f);
  u += 0x7fffu + ((u >> 16) & 1u);     // round-to-nearest-even
  return (unsigned short)(u >> 16);
}

#define WMMA_BF16(A, B, C) \
  __builtin_amdgcn_wmma_f32_16x16x32_bf16(false, (A), false, (B), (short)0, (C), false, false)

// Trunk dims
#define TN 8
#define TC 128
#define THW 16384          // 128*128
#define TTOT 16777216      // 8*128*16384

// ---------------------------------------------------------------------------
// conv3x3 (pad 1, stride 1), 128->128 over [8,128,128,128], implicit GEMM.
// Block: (n, h, co-tile of 32).  grid = 8*128*4 = 4096 blocks.
// ---------------------------------------------------------------------------
__global__ __launch_bounds__(256) void k_conv3x3_wmma(
    const float* __restrict__ x, const float* __restrict__ w,
    const float* __restrict__ bias, float* __restrict__ out)
{
  __shared__ __align__(16) unsigned short Xs[3][130][32];  // [kh][w+1][ci] bf16
  __shared__ __align__(16) unsigned short Ws[9][32][32];   // [kh*3+kw][co][ci]

  const int tid  = threadIdx.x;
  const int lane = tid & 31;
  const int wid  = tid >> 5;
  const int ln   = lane & 15;
  const int half = lane >> 4;

  const int b   = blockIdx.x;
  const int co0 = (b & 3) * 32;
  const int h   = (b >> 2) & 127;
  const int n   = b >> 9;

  const int co_sub = wid & 1;       // which 16-row co sub-tile
  const int ts0 = wid >> 1;         // token sub-tiles (16 tokens each)
  const int ts1 = ts0 + 4;

  v8f acc0 = {0.f,0.f,0.f,0.f,0.f,0.f,0.f,0.f};
  v8f acc1 = acc0;

  for (int cc = 0; cc < 4; ++cc) {          // 4 chunks of 32 input channels
    __syncthreads();
    // stage activations: 3 input rows x 130 (halo) x 32 ci, token-major in LDS
    for (int idx = tid; idx < 3 * 130 * 32; idx += 256) {
      int wi  = idx % 130;
      int tmp = idx / 130;
      int ci  = tmp & 31;
      int kh  = tmp >> 5;
      int hi  = h + kh - 1;
      int wn  = wi - 1;
      float v = 0.f;
      if ((unsigned)hi < 128u && (unsigned)wn < 128u)
        v = x[(((size_t)(n * 128 + cc * 32 + ci)) * 128 + hi) * 128 + wn];
      Xs[kh][wi][ci] = f2bf(v);
    }
    // stage weights: w is [Cout=128][Cin=128][3][3]
    for (int idx = tid; idx < 9 * 32 * 32; idx += 256) {
      int r   = idx % 9;
      int tmp = idx / 9;
      int ci  = tmp & 31;
      int co  = tmp >> 5;
      float v = w[((size_t)((co0 + co) * 128 + cc * 32 + ci)) * 9 + r];
      Ws[r][co][ci] = f2bf(v);
    }
    __syncthreads();

    for (int kh = 0; kh < 3; ++kh) {
      for (int kw = 0; kw < 3; ++kw) {
        FragAB A, B0, B1;
        // A fragment (weights): a[j] -> ci = half*16 + j (contiguous 32B)
        const unsigned short* ap = &Ws[kh * 3 + kw][co_sub * 16 + ln][0];
        A.u[0] = *reinterpret_cast<const uint4*>(ap + half * 16);
        A.u[1] = *reinterpret_cast<const uint4*>(ap + half * 16 + 8);
        // B fragment (activations): b[j] -> ci = half*16 + j (contiguous 32B)
        const unsigned short* bp0 = &Xs[kh][ts0 * 16 + ln + kw][half * 16];
        B0.u[0] = *reinterpret_cast<const uint4*>(bp0);
        B0.u[1] = *reinterpret_cast<const uint4*>(bp0 + 8);
        const unsigned short* bp1 = &Xs[kh][ts1 * 16 + ln + kw][half * 16];
        B1.u[0] = *reinterpret_cast<const uint4*>(bp1);
        B1.u[1] = *reinterpret_cast<const uint4*>(bp1 + 8);
        acc0 = WMMA_BF16(A.v, B0.v, acc0);
        acc1 = WMMA_BF16(A.v, B1.v, acc1);
      }
    }
  }

  // epilogue: C/D layout VGPR r -> M = r + 8*half, N = ln
  for (int r = 0; r < 8; ++r) {
    int co = co0 + co_sub * 16 + r + half * 8;
    float bv = bias[co];
    size_t base = (((size_t)(n * 128 + co)) * 128 + h) * 128;
    out[base + ts0 * 16 + ln] = acc0[r] + bv;
    out[base + ts1 * 16 + ln] = acc1[r] + bv;
  }
}

// ---------------------------------------------------------------------------
// Generic 1x1-conv GEMM: out[M,t] = A[M,K] * B[K,t] + bias,  t over 8*16384
// tokens in NCHW layout (channel stride 16384 per image, K = Cin, M = Cout).
// act: 0 none, 1 relu, 2 sigmoid.  grid = (1024 token tiles, ceil(M/32)).
// ---------------------------------------------------------------------------
__global__ __launch_bounds__(256) void k_gemm_wmma(
    const float* __restrict__ A, const float* __restrict__ B,
    const float* __restrict__ bias, float* __restrict__ out,
    int M, int K, int act)
{
  __shared__ __align__(16) unsigned short As[32][32];
  __shared__ __align__(16) unsigned short Bs[128][32];

  const int tid  = threadIdx.x;
  const int lane = tid & 31;
  const int wid  = tid >> 5;
  const int ln   = lane & 15;
  const int half = lane >> 4;

  const int tile = blockIdx.x;
  const int m0   = blockIdx.y * 32;
  const int n    = (tile * 128) >> 14;
  const int hw0  = (tile * 128) & 16383;

  const int co_sub = wid & 1;
  const int ts0 = wid >> 1;
  const int ts1 = ts0 + 4;

  v8f acc0 = {0.f,0.f,0.f,0.f,0.f,0.f,0.f,0.f};
  v8f acc1 = acc0;

  const int nk = (K + 31) >> 5;
  for (int kc = 0; kc < nk; ++kc) {
    __syncthreads();
    for (int idx = tid; idx < 32 * 32; idx += 256) {
      int mr = idx >> 5, kk = idx & 31;
      int k = kc * 32 + kk;
      float v = (m0 + mr < M && k < K) ? A[(size_t)(m0 + mr) * K + k] : 0.f;
      As[mr][kk] = f2bf(v);
    }
    for (int idx = tid; idx < 128 * 32; idx += 256) {
      int tw = idx & 127, kk = idx >> 7;
      int k = kc * 32 + kk;
      float v = (k < K) ? B[((size_t)(n * K + k)) * 16384 + hw0 + tw] : 0.f;
      Bs[tw][kk] = f2bf(v);
    }
    __syncthreads();

    FragAB Af, B0, B1;
    const unsigned short* ap = &As[co_sub * 16 + ln][0];
    Af.u[0] = *reinterpret_cast<const uint4*>(ap + half * 16);
    Af.u[1] = *reinterpret_cast<const uint4*>(ap + half * 16 + 8);
    const unsigned short* bp0 = &Bs[ts0 * 16 + ln][half * 16];
    B0.u[0] = *reinterpret_cast<const uint4*>(bp0);
    B0.u[1] = *reinterpret_cast<const uint4*>(bp0 + 8);
    const unsigned short* bp1 = &Bs[ts1 * 16 + ln][half * 16];
    B1.u[0] = *reinterpret_cast<const uint4*>(bp1);
    B1.u[1] = *reinterpret_cast<const uint4*>(bp1 + 8);
    acc0 = WMMA_BF16(Af.v, B0.v, acc0);
    acc1 = WMMA_BF16(Af.v, B1.v, acc1);
  }

  for (int r = 0; r < 8; ++r) {
    int co = m0 + co_sub * 16 + r + half * 8;
    if (co < M) {
      float bv = bias[co];
      float v0 = acc0[r] + bv;
      float v1 = acc1[r] + bv;
      if (act == 1) { v0 = fmaxf(v0, 0.f); v1 = fmaxf(v1, 0.f); }
      else if (act == 2) { v0 = 1.f / (1.f + __expf(-v0)); v1 = 1.f / (1.f + __expf(-v1)); }
      size_t base = ((size_t)(n * M + co)) * 16384 + hw0;
      out[base + ts0 * 16 + ln] = v0;
      out[base + ts1 * 16 + ln] = v1;
    }
  }
}

// ---------------------------------------------------------------------------
// Window attention: qkv [8,384,128,128], window = 16x16 (8x8 windows/image).
// Block = (window, 16-query-row tile). S = Q^T K (WMMA) -> softmax (LDS) ->
// O = V P^T (WMMA). grid = (512, 16).
// ---------------------------------------------------------------------------
__global__ __launch_bounds__(256) void k_attn(
    const float* __restrict__ qkv, float* __restrict__ o)
{
  __shared__ __align__(16) float Sl[16][256];
  __shared__ __align__(16) unsigned short Pl[16][256];
  __shared__ float red[16][16];
  __shared__ float rowv[16];

  const int tid  = threadIdx.x;
  const int lane = tid & 31;
  const int wid  = tid >> 5;
  const int ln   = lane & 15;
  const int half = lane >> 4;

  const int win = blockIdx.x;
  const int qt  = blockIdx.y;
  const int n   = win >> 6;
  const int wh  = (win >> 3) & 7;
  const int wwi = win & 7;

  // ----- phase 1: S[16q][256k], each wave handles 2 k-subtiles -----
  {
    const int ks0 = wid * 2, ks1 = wid * 2 + 1;
    v8f s0 = {0.f,0.f,0.f,0.f,0.f,0.f,0.f,0.f};
    v8f s1 = s0;
    // q token (qt,ln); k tokens (ks,ln)
    size_t qb  = ((size_t)(n * 384)) * 16384 + (size_t)(wh * 16 + qt) * 128 + wwi * 16 + ln;
    size_t kb0 = ((size_t)(n * 384 + 128)) * 16384 + (size_t)(wh * 16 + ks0) * 128 + wwi * 16 + ln;
    size_t kb1 = ((size_t)(n * 384 + 128)) * 16384 + (size_t)(wh * 16 + ks1) * 128 + wwi * 16 + ln;
    for (int cc = 0; cc < 4; ++cc) {
      FragAB A, B0, B1;
      for (int j = 0; j < 16; ++j) {
        int ca = cc * 32 + half * 16 + j;                           // A k-pattern
        int cb = cc * 32 + half * 16 + j;                           // B k-pattern
        A.s[j]  = f2bf(qkv[qb  + (size_t)ca * 16384]);
        B0.s[j] = f2bf(qkv[kb0 + (size_t)cb * 16384]);
        B1.s[j] = f2bf(qkv[kb1 + (size_t)cb * 16384]);
      }
      s0 = WMMA_BF16(A.v, B0.v, s0);
      s1 = WMMA_BF16(A.v, B1.v, s1);
    }
    for (int r = 0; r < 8; ++r) {
      Sl[r + half * 8][ks0 * 16 + ln] = s0[r];
      Sl[r + half * 8][ks1 * 16 + ln] = s1[r];
    }
  }
  __syncthreads();

  // ----- softmax over 16 rows of 256 -----
  {
    int row = tid >> 4, col = tid & 15;
    float m = -1e30f;
    for (int j = 0; j < 16; ++j) m = fmaxf(m, Sl[row][col + j * 16]);
    red[row][col] = m;
    __syncthreads();
    if (tid < 16) {
      float mm = -1e30f;
      for (int j = 0; j < 16; ++j) mm = fmaxf(mm, red[tid][j]);
      rowv[tid] = mm;
    }
    __syncthreads();
    float rm = rowv[row];
    float s = 0.f;
    for (int j = 0; j < 16; ++j) {
      float e = __expf(Sl[row][col + j * 16] - rm);
      Sl[row][col + j * 16] = e;
      s += e;
    }
    red[row][col] = s;
    __syncthreads();
    if (tid < 16) {
      float ss = 0.f;
      for (int j = 0; j < 16; ++j) ss += red[tid][j];
      rowv[tid] = ss;
    }
    __syncthreads();
    float inv = 1.f / rowv[row];
    for (int j = 0; j < 16; ++j)
      Pl[row][col + j * 16] = f2bf(Sl[row][col + j * 16] * inv);
  }
  __syncthreads();

  // ----- phase 2: O[c][q] = sum_k V[c][k] P[q][k]; wave = 16-channel tile --
  {
    const int cs = wid;
    v8f acc = {0.f,0.f,0.f,0.f,0.f,0.f,0.f,0.f};
    int cch = cs * 16 + ln;   // A row = channel
    size_t vb = ((size_t)(n * 384 + 256 + cch)) * 16384 + (size_t)(wh * 16) * 128 + wwi * 16;
    for (int kk = 0; kk < 8; ++kk) {
      FragAB A, B;
      for (int j = 0; j < 16; ++j) {
        int tk = kk * 32 + half * 16 + j;
        A.s[j] = f2bf(qkv[vb + (size_t)(tk >> 4) * 128 + (tk & 15)]);
      }
      const unsigned short* bp = &Pl[ln][kk * 32 + half * 16];
      B.u[0] = *reinterpret_cast<const uint4*>(bp);
      B.u[1] = *reinterpret_cast<const uint4*>(bp + 8);
      acc = WMMA_BF16(A.v, B.v, acc);
    }
    for (int r = 0; r < 8; ++r) {
      int ch = cs * 16 + r + half * 8;
      o[((size_t)(n * 128 + ch)) * 16384 + (size_t)(wh * 16 + qt) * 128 + wwi * 16 + ln] = acc[r];
    }
  }
}

// ---------------------------------------------------------------------------
// BatchNorm (training stats): per-channel mean/rstd, then normalize (+relu).
// ---------------------------------------------------------------------------
__global__ __launch_bounds__(256) void k_bnstats(
    const float* __restrict__ y, float* __restrict__ st)
{
  __shared__ float sh1[256], sh2[256];
  const int c = blockIdx.x, tid = threadIdx.x;
  float a = 0.f, b = 0.f;
  for (int n = 0; n < TN; ++n) {
    const float* p = y + ((size_t)(n * TC + c)) * THW;
    for (int i = tid; i < THW; i += 256) {
      float v = p[i];
      a += v; b += v * v;
    }
  }
  sh1[tid] = a; sh2[tid] = b;
  __syncthreads();
  for (int o = 128; o > 0; o >>= 1) {
    if (tid < o) { sh1[tid] += sh1[tid + o]; sh2[tid] += sh2[tid + o]; }
    __syncthreads();
  }
  if (tid == 0) {
    float mu  = sh1[0] * (1.f / 131072.f);
    float var = sh2[0] * (1.f / 131072.f) - mu * mu;
    st[c] = mu;
    st[TC + c] = rsqrtf(var + 1e-5f);
  }
}

__global__ __launch_bounds__(256) void k_bnapply(
    float* __restrict__ y, const float* __restrict__ st,
    const float* __restrict__ g, const float* __restrict__ b, int relu)
{
  int idx = blockIdx.x * 256 + threadIdx.x;
  if (idx >= TTOT) return;
  int c = (idx >> 14) & 127;
  float v = g[c] * (y[idx] - st[c]) * st[TC + c] + b[c];
  if (relu) v = fmaxf(v, 0.f);
  y[idx] = v;
}

__global__ __launch_bounds__(256) void k_mul(
    float* __restrict__ dst, const float* __restrict__ a, const float* __restrict__ b)
{
  int idx = blockIdx.x * 256 + threadIdx.x;
  if (idx < TTOT) dst[idx] = a[idx] * b[idx];
}

__global__ __launch_bounds__(256) void k_add3(
    float* __restrict__ h, const float* __restrict__ p, const float* __restrict__ t)
{
  int idx = blockIdx.x * 256 + threadIdx.x;
  if (idx < TTOT) h[idx] = h[idx] + p[idx] + t[idx];
}

// ---------------------------------------------------------------------------
// Encoder: reflect-pad(1), 4x4 conv stride 2, relu.  Thread per output.
// ---------------------------------------------------------------------------
__device__ __forceinline__ int refl(int i, int H) {
  if (i < 0) i = -i;
  if (i >= H) i = 2 * H - 2 - i;
  return i;
}

__global__ __launch_bounds__(256) void k_enc(
    const float* __restrict__ in, const float* __restrict__ w,
    const float* __restrict__ b, float* __restrict__ out,
    int Cin, int Cout, int Hin, int Hout)
{
  int idx = blockIdx.x * 256 + threadIdx.x;
  int total = TN * Cout * Hout * Hout;
  if (idx >= total) return;
  int ow = idx % Hout;
  int oh = (idx / Hout) % Hout;
  int co = (idx / (Hout * Hout)) % Cout;
  int n  = idx / (Hout * Hout * Cout);
  float sum = b[co];
  for (int ci = 0; ci < Cin; ++ci) {
    const float* xp = in + ((size_t)(n * Cin + ci)) * Hin * Hin;
    const float* wp = w + ((size_t)(co * Cin + ci)) * 16;
    for (int kh = 0; kh < 4; ++kh) {
      int ih = refl(oh * 2 - 1 + kh, Hin);
      for (int kw = 0; kw < 4; ++kw) {
        int iw = refl(ow * 2 - 1 + kw, Hin);
        sum += xp[(size_t)ih * Hin + iw] * wp[kh * 4 + kw];
      }
    }
  }
  out[idx] = fmaxf(sum, 0.f);
}

// ---------------------------------------------------------------------------
// Decoder: conv_transpose stride 2, kernel 4.  dec1: 128->64, relu.
// w layout [Cin][Cout][4][4]; effective w_t[o][i][k] = w[i][o][3-k].
// ---------------------------------------------------------------------------
__global__ __launch_bounds__(256) void k_dec1(
    const float* __restrict__ in, const float* __restrict__ w,
    const float* __restrict__ b, float* __restrict__ out)
{
  int idx = blockIdx.x * 256 + threadIdx.x;
  if (idx >= TN * 64 * 256 * 256) return;
  int ow = idx % 256;
  int oh = (idx / 256) % 256;
  int co = (idx / 65536) % 64;
  int n  = idx / (65536 * 64);
  float sum = b[co];
  for (int kh = 0; kh < 4; ++kh) {
    int qh = oh - 2 + kh;
    if (qh < 0 || qh > 254 || (qh & 1)) continue;
    int ih = qh >> 1;
    for (int kw = 0; kw < 4; ++kw) {
      int qw = ow - 2 + kw;
      if (qw < 0 || qw > 254 || (qw & 1)) continue;
      int iw = qw >> 1;
      const float* xp = in + ((size_t)(n * 128)) * THW + (size_t)ih * 128 + iw;
      const float* wp = w + ((3 - kh) * 4 + (3 - kw)) + (size_t)co * 16;
      for (int ci = 0; ci < 128; ++ci)
        sum += xp[(size_t)ci * THW] * wp[(size_t)ci * 1024];
    }
  }
  out[idx] = fmaxf(sum, 0.f);
}

__global__ __launch_bounds__(256) void k_dec2(
    const float* __restrict__ in, const float* __restrict__ w,
    const float* __restrict__ b, float* __restrict__ out)
{
  int idx = blockIdx.x * 256 + threadIdx.x;
  if (idx >= TN * 3 * 512 * 512) return;
  int ow = idx % 512;
  int oh = (idx / 512) % 512;
  int co = (idx / 262144) % 3;
  int n  = idx / 786432;
  float sum = b[co];
  for (int kh = 0; kh < 4; ++kh) {
    int qh = oh - 2 + kh;
    if (qh < 0 || qh > 510 || (qh & 1)) continue;
    int ih = qh >> 1;
    for (int kw = 0; kw < 4; ++kw) {
      int qw = ow - 2 + kw;
      if (qw < 0 || qw > 510 || (qw & 1)) continue;
      int iw = qw >> 1;
      const float* xp = in + ((size_t)(n * 64)) * 65536 + (size_t)ih * 256 + iw;
      const float* wp = w + ((3 - kh) * 4 + (3 - kw)) + (size_t)co * 16;
      for (int ci = 0; ci < 64; ++ci)
        sum += xp[(size_t)ci * 65536] * wp[(size_t)ci * 48];
    }
  }
  out[idx] = tanhf(sum);
}

// ---------------------------------------------------------------------------
extern "C" void kernel_launch(void* const* d_in, const int* in_sizes, int n_in,
                              void* d_out, int out_size, void* d_ws, size_t ws_size,
                              hipStream_t stream) {
  (void)in_sizes; (void)n_in; (void)out_size; (void)ws_size;
  const float* x       = (const float*)d_in[0];
  const float* enc_w1  = (const float*)d_in[1];
  const float* enc_b1  = (const float*)d_in[2];
  const float* enc_w2  = (const float*)d_in[3];
  const float* enc_b2  = (const float*)d_in[4];
  const float* conv1_w = (const float*)d_in[5];
  const float* conv1_b = (const float*)d_in[6];
  const float* bn1_g   = (const float*)d_in[7];
  const float* bn1_b   = (const float*)d_in[8];
  const float* conv2_w = (const float*)d_in[9];
  const float* conv2_b = (const float*)d_in[10];
  const float* bn2_g   = (const float*)d_in[11];
  const float* bn2_b   = (const float*)d_in[12];
  const float* attn1_w = (const float*)d_in[13];
  const float* attn1_b = (const float*)d_in[14];
  const float* attn2_w = (const float*)d_in[15];
  const float* attn2_b = (const float*)d_in[16];
  const float* qkv_w   = (const float*)d_in[17];
  const float* qkv_b   = (const float*)d_in[18];
  const float* proj_w  = (const float*)d_in[19];
  const float* proj_b  = (const float*)d_in[20];
  const float* dec_w1  = (const float*)d_in[21];
  const float* dec_b1  = (const float*)d_in[22];
  const float* dec_w2  = (const float*)d_in[23];
  const float* dec_b2  = (const float*)d_in[24];
  float* out = (float*)d_out;

  const size_t T = (size_t)TTOT;        // 16,777,216 floats (67 MB)
  float* ws   = (float*)d_ws;
  float* bh   = ws;                     // h
  float* by   = ws + T;                 // conv1/bn1 out; later proj out
  float* bc   = ws + 2 * T;             // conv2/bn2 out
  float* bt   = ws + 3 * T;             // gated input (c * gate)
  float* bo   = ws + 4 * T;             // gate; later attention output
  float* big  = ws + 5 * T;             // qkv (3T); also enc1/dec1 (2T)
  float* ba   = ws + 8 * T;             // attn1 activations (2M floats)
  float* st   = ws + 8 * T + 2097152;   // BN stats (256 floats)

  // ---- encoder ----
  k_enc<<<131072, 256, 0, stream>>>(x, enc_w1, enc_b1, big, 3, 64, 512, 256);
  k_enc<<<65536, 256, 0, stream>>>(big, enc_w2, enc_b2, bh, 64, 128, 256, 128);

  // ---- 12 residual attention blocks ----
  for (int i = 0; i < 12; ++i) {
    const float* w1 = conv1_w + (size_t)i * 147456;
    const float* b1 = conv1_b + (size_t)i * 128;
    const float* w2 = conv2_w + (size_t)i * 147456;
    const float* b2 = conv2_b + (size_t)i * 128;

    k_conv3x3_wmma<<<4096, 256, 0, stream>>>(bh, w1, b1, by);
    k_bnstats<<<128, 256, 0, stream>>>(by, st);
    k_bnapply<<<65536, 256, 0, stream>>>(by, st, bn1_g + i * 128, bn1_b + i * 128, 1);

    k_conv3x3_wmma<<<4096, 256, 0, stream>>>(by, w2, b2, bc);
    k_bnstats<<<128, 256, 0, stream>>>(bc, st);
    k_bnapply<<<65536, 256, 0, stream>>>(bc, st, bn2_g + i * 128, bn2_b + i * 128, 0);

    // gate = sigmoid(attn2(relu(attn1(c))))
    k_gemm_wmma<<<dim3(1024, 1), 256, 0, stream>>>(
        attn1_w + (size_t)i * 2048, bc, attn1_b + i * 16, ba, 16, 128, 1);
    k_gemm_wmma<<<dim3(1024, 4), 256, 0, stream>>>(
        attn2_w + (size_t)i * 2048, ba, attn2_b + i * 128, bo, 128, 16, 2);
    k_mul<<<65536, 256, 0, stream>>>(bt, bc, bo);          // t = c * gate

    // window attention on t
    k_gemm_wmma<<<dim3(1024, 12), 256, 0, stream>>>(
        qkv_w + (size_t)i * 49152, bt, qkv_b + i * 384, big, 384, 128, 0);
    k_attn<<<dim3(512, 16), 256, 0, stream>>>(big, bo);
    k_gemm_wmma<<<dim3(1024, 4), 256, 0, stream>>>(
        proj_w + (size_t)i * 16384, bo, proj_b + i * 128, by, 128, 128, 0);

    k_add3<<<65536, 256, 0, stream>>>(bh, by, bt);         // h += proj + t
  }

  // ---- decoder ----
  k_dec1<<<131072, 256, 0, stream>>>(bh, dec_w1, dec_b1, big);
  k_dec2<<<24576, 256, 0, stream>>>(big, dec_w2, dec_b2, out);
}
